// CrossMamba__44787918963233
// MI455X (gfx1250) — matrix operand order repaired and checked by reading, not compiled, not run on hardware
//
#include <hip/hip_runtime.h>
#include <math.h>

#define DIMC    384
#define D_INNER 768
#define LSEQ    4096
#define B_SZ    4
#define HWDIM   64
#define NROWS   (B_SZ * LSEQ)   // 16384

typedef __bf16 bf16_t;
typedef bf16_t v16bf __attribute__((ext_vector_type(16)));
typedef float  v8f   __attribute__((ext_vector_type(8)));

__device__ __forceinline__ unsigned short f2bf(float f) {
  union { float f; unsigned u; } x; x.f = f;
  unsigned r = x.u + 0x7fffu + ((x.u >> 16) & 1u);   // round-to-nearest-even
  return (unsigned short)(r >> 16);
}

// ---------------------------------------------------------------------------
// WMMA bf16 GEMM:  C[M,N](f32) = A[M,K](bf16) * Bt[N,K](bf16)^T (+bias[N])
// Block tile 128x128, 8 wave32 waves (4x2), wave tile 32x64 (8 wmma/K-step).
// Double-buffered LDS, software-pipelined global->LDS staging, 1 barrier/step.
// ---------------------------------------------------------------------------
__global__ __launch_bounds__(256) void k_gemm_bf16(
    const unsigned short* __restrict__ Ag, const unsigned short* __restrict__ Bg,
    const float* __restrict__ bias, float* __restrict__ Cg,
    int M, int N, int K)
{
  __shared__ unsigned short lA[2][128 * 32];
  __shared__ unsigned short lB[2][128 * 32];
  const int tid  = threadIdx.x;
  const int lane = tid & 31;
  const int wave = tid >> 5;
  const int wm   = wave >> 1;          // 0..3 -> 32-row band
  const int wn   = wave & 1;           // 0..1 -> 64-col band
  const int row0 = blockIdx.y * 128;
  const int n0   = blockIdx.x * 128;
  const int hl   = lane >> 4;          // lane half
  const int lr   = lane & 15;

  // staging: each thread owns two 16B chunks of A panel and two of B panel
  const int sr = tid >> 2;             // 0..63 (panel row)
  const int sc = (tid & 3) << 3;       // element col: 0,8,16,24
  const unsigned short* aP0 = Ag + (size_t)(row0 + sr) * K + sc;
  const unsigned short* aP1 = aP0 + (size_t)64 * K;
  const bool bv0 = (n0 + sr)      < N;
  const bool bv1 = (n0 + sr + 64) < N;
  const unsigned short* bP0 = Bg + (size_t)(n0 + sr) * K + sc;
  const unsigned short* bP1 = bP0 + (size_t)64 * K;
  const int sOfs0 = sr * 32 + sc;
  const int sOfs1 = sOfs0 + 64 * 32;

  uint4 ra0, ra1;
  uint4 rb0 = make_uint4(0u, 0u, 0u, 0u), rb1 = make_uint4(0u, 0u, 0u, 0u);

  // prologue: stage panel 0
  ra0 = *(const uint4*)aP0;
  ra1 = *(const uint4*)aP1;
  if (bv0) rb0 = *(const uint4*)bP0;
  if (bv1) rb1 = *(const uint4*)bP1;
  *(uint4*)(&lA[0][sOfs0]) = ra0;
  *(uint4*)(&lA[0][sOfs1]) = ra1;
  *(uint4*)(&lB[0][sOfs0]) = rb0;
  *(uint4*)(&lB[0][sOfs1]) = rb1;
  __syncthreads();

  v8f acc[2][4];
  for (int i = 0; i < 2; ++i)
    for (int j = 0; j < 4; ++j)
      for (int e = 0; e < 8; ++e) acc[i][j][e] = 0.f;

  const int nk = K >> 5;
  for (int i = 0; i < nk; ++i) {
    const int cur = i & 1;
    const bool more = (i + 1) < nk;
    if (more) {               // issue next panel's global loads (overlap wmma)
      aP0 += 32; aP1 += 32; bP0 += 32; bP1 += 32;
      ra0 = *(const uint4*)aP0;
      ra1 = *(const uint4*)aP1;
      if (bv0) rb0 = *(const uint4*)bP0;
      if (bv1) rb1 = *(const uint4*)bP1;
    }

    // fragments per ISA 7.12.2 layouts (wave32), b128 LDS reads
    union FU { v16bf v; uint4 q[2]; };
    FU a[2], b[4];
    #pragma unroll
    for (int ii = 0; ii < 2; ++ii) {
      const unsigned short* base = &lA[cur][(wm * 32 + ii * 16 + lr) * 32];
      a[ii].q[0] = *(const uint4*)(base + hl * 8);        // K 0..7   (+8*hl)
      a[ii].q[1] = *(const uint4*)(base + 16 + hl * 8);   // K 16..23 (+8*hl)
    }
    #pragma unroll
    for (int jj = 0; jj < 4; ++jj) {
      const unsigned short* base = &lB[cur][(wn * 64 + jj * 16 + lr) * 32];
      b[jj].q[0] = *(const uint4*)(base + hl * 8);        // K 0..7   (+8*hl)
      b[jj].q[1] = *(const uint4*)(base + 16 + hl * 8);   // K 16..23 (+8*hl)
    }
    #pragma unroll
    for (int ii = 0; ii < 2; ++ii)
      #pragma unroll
      for (int jj = 0; jj < 4; ++jj)
        acc[ii][jj] = __builtin_amdgcn_wmma_f32_16x16x32_bf16(
            false, a[ii].v, false, b[jj].v, (short)0, acc[ii][jj], false, false);

    if (more) {               // stage next panel into the other buffer
      const int nxt = cur ^ 1;
      *(uint4*)(&lA[nxt][sOfs0]) = ra0;
      *(uint4*)(&lA[nxt][sOfs1]) = ra1;
      *(uint4*)(&lB[nxt][sOfs0]) = rb0;
      *(uint4*)(&lB[nxt][sOfs1]) = rb1;
    }
    __syncthreads();
  }

  // epilogue: C frag VGPR v -> row v + 8*half, col = lane&15
  for (int ii = 0; ii < 2; ++ii) {
    for (int jj = 0; jj < 4; ++jj) {
      int gcol = n0 + wn * 64 + jj * 16 + lr;
      if (gcol >= N) continue;
      float bvs = bias ? bias[gcol] : 0.f;
      #pragma unroll
      for (int v = 0; v < 8; ++v) {
        int grow = row0 + wm * 32 + ii * 16 + v + hl * 8;
        Cg[(size_t)grow * N + gcol] = acc[ii][jj][v] + bvs;
      }
    }
  }
}

// ---------------------------------------------------------------------------
// Elementwise / helper kernels
// ---------------------------------------------------------------------------
__global__ void k_concat_bf(const float* __restrict__ ms, const float* __restrict__ pan,
                            unsigned short* __restrict__ cat, int n)
{
  int i = blockIdx.x * blockDim.x + threadIdx.x;
  if (i >= n) return;
  int row = i / DIMC, c = i % DIMC;
  cat[(size_t)row * 768 + c]        = f2bf(ms[i]);
  cat[(size_t)row * 768 + 384 + c]  = f2bf(pan[i]);
}

// pad/cast f32 -> bf16 with row & column zero padding
__global__ void k_cast_pad(const float* __restrict__ src, unsigned short* __restrict__ dst,
                           int srows, int src_ld, int scols, int drows, int dcols)
{
  int i = blockIdx.x * blockDim.x + threadIdx.x;
  if (i >= drows * dcols) return;
  int r = i / dcols, c = i % dcols;
  float v = (r < srows && c < scols) ? src[(size_t)r * src_ld + c] : 0.f;
  dst[i] = f2bf(v);
}

// one wave32 per row, dim = 384
__global__ __launch_bounds__(256) void k_layernorm_bf16(
    const float* __restrict__ x, const float* __restrict__ w, const float* __restrict__ bsh,
    unsigned short* __restrict__ out, int rows)
{
  int wave = threadIdx.x >> 5, lane = threadIdx.x & 31;
  int row = blockIdx.x * 8 + wave;
  if (row >= rows) return;
  const float* xr = x + (size_t)row * DIMC;
  float s = 0.f, s2 = 0.f;
  #pragma unroll
  for (int c = lane; c < DIMC; c += 32) { float v = xr[c]; s += v; s2 += v * v; }
  #pragma unroll
  for (int m = 16; m >= 1; m >>= 1) { s += __shfl_xor(s, m, 32); s2 += __shfl_xor(s2, m, 32); }
  float mu  = s  * (1.f / DIMC);
  float var = s2 * (1.f / DIMC) - mu * mu;
  float inv = rsqrtf(var + 1e-5f);
  unsigned short* orow = out + (size_t)row * DIMC;
  for (int c = lane; c < DIMC; c += 32)
    orow[c] = f2bf((xr[c] - mu) * inv * w[c] + bsh[c]);
}

// causal depthwise conv1d (K=4) along L + SiLU; optional f32 / bf16 outputs
__global__ void k_dwconv_silu(const float* __restrict__ in, int in_ld,
                              const float* __restrict__ w, const float* __restrict__ b,
                              float* __restrict__ of, unsigned short* __restrict__ ob, int n)
{
  int i = blockIdx.x * blockDim.x + threadIdx.x;
  if (i >= n) return;
  int d = i % D_INNER;
  int rl = i / D_INNER;
  int l = rl % LSEQ;
  float acc = b[d];
  #pragma unroll
  for (int j = 0; j < 4; ++j) {
    int ls = l - 3 + j;
    if (ls >= 0) acc += w[d * 4 + j] * in[(size_t)(rl - l + ls) * in_ld + d];
  }
  float s = acc / (1.f + __expf(-acc));
  if (of) of[i] = s;
  if (ob) ob[i] = f2bf(s);
}

__global__ void k_softplus_inplace(float* __restrict__ x, int n)
{
  int i = blockIdx.x * blockDim.x + threadIdx.x;
  if (i >= n) return;
  float v = x[i];
  x[i] = (v > 20.f) ? v : log1pf(__expf(v));
}

// selective scan, fused with +u*D and y*silu(z); emits bf16 for out_proj GEMM
__global__ __launch_bounds__(256) void k_scan(
    const float* __restrict__ u, const float* __restrict__ dt,
    const float* __restrict__ dbl, const float* __restrict__ cm,
    const float* __restrict__ zsrc, const float* __restrict__ A_log,
    const float* __restrict__ Dp, unsigned short* __restrict__ yg)
{
  int t0 = blockIdx.x * blockDim.x + threadIdx.x;
  if (t0 >= B_SZ * D_INNER) return;
  int b = t0 / D_INNER, d = t0 % D_INNER;
  float a[16], h[16];
  #pragma unroll
  for (int n = 0; n < 16; ++n) { a[n] = -__expf(A_log[d * 16 + n]); h[n] = 0.f; }
  float Dv = Dp[d];
  size_t rb = (size_t)b * LSEQ;
  const float* up  = u    + rb * D_INNER + d;
  const float* dtp = dt   + rb * D_INNER + d;
  const float* zp  = zsrc + rb * 1536 + 768 + d;
  const float* Bp  = dbl  + rb * 48 + 24;
  const float* Cp  = cm   + rb * 16;
  unsigned short* yp = yg + rb * D_INNER + d;
  for (int t = 0; t < LSEQ; ++t) {
    float dtv = dtp[(size_t)t * D_INNER];
    float uv  = up [(size_t)t * D_INNER];
    float du  = dtv * uv;
    float acc = 0.f;
    #pragma unroll
    for (int n = 0; n < 16; ++n) {
      h[n] = __expf(dtv * a[n]) * h[n] + du * Bp[(size_t)t * 48 + n];
      acc += h[n] * Cp[(size_t)t * 16 + n];
    }
    float y  = acc + uv * Dv;
    float zv = zp[(size_t)t * 1536];
    float g  = zv / (1.f + __expf(-zv));
    yp[(size_t)t * D_INNER] = f2bf(y * g);
  }
}

__global__ void k_add(float* __restrict__ a, const float* __restrict__ b, int n)
{
  int i = blockIdx.x * blockDim.x + threadIdx.x;
  if (i < n) a[i] += b[i];
}

// 3x3 depthwise conv2d SAME on (B, C, 64, 64); gf is [b][l][c] row-major
__global__ void k_dwconv2d(const float* __restrict__ gf, const float* __restrict__ w,
                           const float* __restrict__ b, float* __restrict__ out)
{
  int i = blockIdx.x * blockDim.x + threadIdx.x;
  if (i >= B_SZ * DIMC * HWDIM * HWDIM) return;
  int x = i % HWDIM; int t = i / HWDIM;
  int y = t % HWDIM; t /= HWDIM;
  int c = t % DIMC;  int bt = t / DIMC;
  float acc = b[c];
  #pragma unroll
  for (int ky = 0; ky < 3; ++ky) {
    int yy = y + ky - 1; if (yy < 0 || yy >= HWDIM) continue;
    #pragma unroll
    for (int kx = 0; kx < 3; ++kx) {
      int xx = x + kx - 1; if (xx < 0 || xx >= HWDIM) continue;
      acc += w[c * 9 + ky * 3 + kx] *
             gf[((size_t)bt * LSEQ + yy * HWDIM + xx) * DIMC + c];
    }
  }
  out[i] = acc;
}

// ---------------------------------------------------------------------------
extern "C" void kernel_launch(void* const* d_in, const int* in_sizes, int n_in,
                              void* d_out, int out_size, void* d_ws, size_t ws_size,
                              hipStream_t stream)
{
  (void)in_sizes; (void)n_in; (void)out_size; (void)ws_size;
  const float* ms          = (const float*)d_in[0];
  const float* pan         = (const float*)d_in[1];
  const float* reduce_W    = (const float*)d_in[2];
  const float* reduce_b    = (const float*)d_in[3];
  const float* ln1_w       = (const float*)d_in[4];
  const float* ln1_b       = (const float*)d_in[5];
  const float* ln2_w       = (const float*)d_in[6];
  const float* ln2_b       = (const float*)d_in[7];
  const float* ln3_w       = (const float*)d_in[8];
  const float* ln3_b       = (const float*)d_in[9];
  const float* in_proj_W   = (const float*)d_in[10];
  const float* in_proj_b_W = (const float*)d_in[11];
  const float* in_proj_c_W = (const float*)d_in[12];
  const float* conv_w      = (const float*)d_in[13];
  const float* conv_bias   = (const float*)d_in[14];
  const float* conv_b_w    = (const float*)d_in[15];
  const float* conv_b_bias = (const float*)d_in[16];
  const float* conv_c_w    = (const float*)d_in[17];
  const float* conv_c_bias = (const float*)d_in[18];
  const float* x_proj_W    = (const float*)d_in[19];
  const float* x_proj_c_W  = (const float*)d_in[20];
  const float* dt_proj_W   = (const float*)d_in[21];
  const float* dt_proj_b   = (const float*)d_in[22];
  const float* A_log       = (const float*)d_in[23];
  const float* Dp          = (const float*)d_in[24];
  const float* out_proj_W  = (const float*)d_in[25];
  const float* dwconv_w    = (const float*)d_in[26];
  const float* dwconv_b    = (const float*)d_in[27];
  float* out = (float*)d_out;

  size_t off = 0;
  auto alloc = [&](size_t bytes) -> void* {
    void* p = (char*)d_ws + off;
    off += (bytes + 255) & ~(size_t)255;
    return p;
  };
  unsigned short* cat_bf    = (unsigned short*)alloc((size_t)NROWS * 768 * 2);
  unsigned short* msn_bf    = (unsigned short*)alloc((size_t)NROWS * 384 * 2);
  unsigned short* pann_bf   = (unsigned short*)alloc((size_t)NROWS * 384 * 2);
  unsigned short* conn_bf   = (unsigned short*)alloc((size_t)NROWS * 384 * 2);
  float*          concat_f  = (float*)alloc((size_t)NROWS * 384 * 4);
  float*          xz        = (float*)alloc((size_t)NROWS * 1536 * 4);
  float*          xb_in     = (float*)alloc((size_t)NROWS * 768 * 4);  // later reused as dt
  float*          xc_in     = (float*)alloc((size_t)NROWS * 768 * 4);
  float*          x_f32     = (float*)alloc((size_t)NROWS * 768 * 4);
  unsigned short* xb_bf     = (unsigned short*)alloc((size_t)NROWS * 768 * 2);
  unsigned short* xc_bf     = (unsigned short*)alloc((size_t)NROWS * 768 * 2);
  float*          dbl       = (float*)alloc((size_t)NROWS * 48 * 4);
  unsigned short* dbl_bf    = (unsigned short*)alloc((size_t)NROWS * 32 * 2);
  float*          cm        = (float*)alloc((size_t)NROWS * 16 * 4);
  unsigned short* yg_bf     = (unsigned short*)alloc((size_t)NROWS * 768 * 2);
  float*          gf        = (float*)alloc((size_t)NROWS * 384 * 4);
  unsigned short* reduce_bfW  = (unsigned short*)alloc((size_t)384 * 768 * 2);
  unsigned short* inproj_bfW  = (unsigned short*)alloc((size_t)1536 * 384 * 2);
  unsigned short* bproj_bfW   = (unsigned short*)alloc((size_t)768 * 384 * 2);
  unsigned short* cproj_bfW   = (unsigned short*)alloc((size_t)768 * 384 * 2);
  unsigned short* xprojp_bfW  = (unsigned short*)alloc((size_t)48 * 768 * 2);
  unsigned short* xprojc_bfW  = (unsigned short*)alloc((size_t)16 * 768 * 2);
  unsigned short* dtprojp_bfW = (unsigned short*)alloc((size_t)768 * 32 * 2);
  unsigned short* outproj_bfW = (unsigned short*)alloc((size_t)384 * 768 * 2);

  const int TB = 256;
  auto grid1 = [&](int n) { return dim3((unsigned)((n + TB - 1) / TB)); };

  // weight casts (with zero padding where shapes aren't WMMA-friendly)
  k_cast_pad<<<grid1(384 * 768),  TB, 0, stream>>>(reduce_W,    reduce_bfW,  384, 768, 768, 384, 768);
  k_cast_pad<<<grid1(1536 * 384), TB, 0, stream>>>(in_proj_W,   inproj_bfW, 1536, 384, 384, 1536, 384);
  k_cast_pad<<<grid1(768 * 384),  TB, 0, stream>>>(in_proj_b_W, bproj_bfW,   768, 384, 384, 768, 384);
  k_cast_pad<<<grid1(768 * 384),  TB, 0, stream>>>(in_proj_c_W, cproj_bfW,   768, 384, 384, 768, 384);
  k_cast_pad<<<grid1(48 * 768),   TB, 0, stream>>>(x_proj_W,    xprojp_bfW,   40, 768, 768, 48, 768);
  k_cast_pad<<<grid1(16 * 768),   TB, 0, stream>>>(x_proj_c_W,  xprojc_bfW,   16, 768, 768, 16, 768);
  k_cast_pad<<<grid1(768 * 32),   TB, 0, stream>>>(dt_proj_W,   dtprojp_bfW, 768,  24,  24, 768, 32);
  k_cast_pad<<<grid1(384 * 768),  TB, 0, stream>>>(out_proj_W,  outproj_bfW, 384, 768, 768, 384, 768);

  // concat (bf16) + reduce GEMM + layernorms
  k_concat_bf<<<grid1(NROWS * 384), TB, 0, stream>>>(ms, pan, cat_bf, NROWS * 384);
  k_gemm_bf16<<<dim3(3, NROWS / 128), TB, 0, stream>>>(cat_bf, reduce_bfW, reduce_b, concat_f, NROWS, 384, 768);
  k_layernorm_bf16<<<dim3(NROWS / 8), TB, 0, stream>>>(ms,       ln1_w, ln1_b, msn_bf,  NROWS);
  k_layernorm_bf16<<<dim3(NROWS / 8), TB, 0, stream>>>(pan,      ln2_w, ln2_b, pann_bf, NROWS);
  k_layernorm_bf16<<<dim3(NROWS / 8), TB, 0, stream>>>(concat_f, ln3_w, ln3_b, conn_bf, NROWS);

  // input projections
  k_gemm_bf16<<<dim3(12, NROWS / 128), TB, 0, stream>>>(msn_bf,  inproj_bfW, nullptr, xz,    NROWS, 1536, 384);
  k_gemm_bf16<<<dim3(6,  NROWS / 128), TB, 0, stream>>>(pann_bf, bproj_bfW,  nullptr, xb_in, NROWS, 768, 384);
  k_gemm_bf16<<<dim3(6,  NROWS / 128), TB, 0, stream>>>(conn_bf, cproj_bfW,  nullptr, xc_in, NROWS, 768, 384);

  // causal depthwise conv + SiLU
  int nconv = NROWS * 768;
  k_dwconv_silu<<<grid1(nconv), TB, 0, stream>>>(xz,   1536, conv_w,   conv_bias,   x_f32,  nullptr, nconv);
  k_dwconv_silu<<<grid1(nconv), TB, 0, stream>>>(xb_in, 768, conv_b_w, conv_b_bias, nullptr, xb_bf,  nconv);
  k_dwconv_silu<<<grid1(nconv), TB, 0, stream>>>(xc_in, 768, conv_c_w, conv_c_bias, nullptr, xc_bf,  nconv);

  // dbl = x_b @ x_proj^T (N padded to 48); dt path; Cm
  k_gemm_bf16<<<dim3(1, NROWS / 128), TB, 0, stream>>>(xb_bf, xprojp_bfW, nullptr, dbl, NROWS, 48, 768);
  k_cast_pad<<<grid1(NROWS * 32), TB, 0, stream>>>(dbl, dbl_bf, NROWS, 48, 32, NROWS, 32);
  float* dt = xb_in;  // reuse dead buffer
  k_gemm_bf16<<<dim3(6, NROWS / 128), TB, 0, stream>>>(dbl_bf, dtprojp_bfW, dt_proj_b, dt, NROWS, 768, 32);
  k_softplus_inplace<<<grid1(nconv), TB, 0, stream>>>(dt, nconv);
  k_gemm_bf16<<<dim3(1, NROWS / 128), TB, 0, stream>>>(xc_bf, xprojc_bfW, nullptr, cm, NROWS, 16, 768);

  // selective scan fused with +u*D and y*silu(z) gate
  k_scan<<<dim3((B_SZ * 768 + TB - 1) / TB), TB, 0, stream>>>(
      x_f32, dt, dbl, cm, xz, A_log, Dp, yg_bf);

  // out projection + residual + final depthwise 3x3
  k_gemm_bf16<<<dim3(3, NROWS / 128), TB, 0, stream>>>(yg_bf, outproj_bfW, nullptr, gf, NROWS, 384, 768);
  k_add<<<grid1(NROWS * 384), TB, 0, stream>>>(gf, ms, NROWS * 384);
  k_dwconv2d<<<grid1(B_SZ * DIMC * HWDIM * HWDIM), TB, 0, stream>>>(gf, dwconv_w, dwconv_b, out);
}
